// LinearSDE_62345745268857
// MI455X (gfx1250) — matrix operand order repaired and checked by reading, not compiled, not run on hardware
//
#include <hip/hip_runtime.h>
#include <hip/hip_bf16.h>
#include <math.h>

// ---------------------------------------------------------------------------
// LinearSDE on MI455X (gfx1250, wave32, WMMA + TDM)
// B=16, T=2048, LD=256, NB=32, BT=32768
// ---------------------------------------------------------------------------

#define BT    32768
#define LD    256
#define NB    32
#define Bsz   16
#define Tlen  2048
#define CHUNK 64
#define NCH   (Tlen / CHUNK)   // 32
#define EPSv  1e-06f

typedef __bf16 bf16_t;
typedef __attribute__((ext_vector_type(16))) __bf16 v16bf;
typedef __attribute__((ext_vector_type(8)))  float  v8f;
typedef __attribute__((ext_vector_type(4))) unsigned int u32x4;
typedef __attribute__((ext_vector_type(8))) int         i32x8;
typedef __attribute__((ext_vector_type(4))) int         i32x4;

// ---- TDM: stage a contiguous rows x cols bf16 slab into LDS -----------------
// 2D descriptor (D#): data_size=2B, tile = (cols, rows), stride = cols.
// Values are wave-uniform (kernel args + blockIdx), so descriptor lives in SGPRs.
__device__ __forceinline__ void tdm_load_2d(bf16_t* lds_dst, const bf16_t* gsrc,
                                            unsigned rows, unsigned cols) {
  unsigned long long ga = (unsigned long long)(size_t)gsrc;
  unsigned lo = (unsigned)(size_t)lds_dst;              // LDS byte address
  u32x4 g0;
  g0[0] = 1u;                                           // count=1, user D#
  g0[1] = lo;                                           // lds_addr
  g0[2] = (unsigned)(ga & 0xffffffffu);                 // global_addr[31:0]
  g0[3] = (unsigned)((ga >> 32) & 0x01ffffffu) | (2u << 30); // addr[56:32] | type=2
  i32x8 g1 = (i32x8)0;
  g1[0] = 0x00010000;                                   // data_size=1 (2 bytes)
  g1[1] = (int)(cols << 16);                            // tensor_dim0[15:0]
  g1[2] = (int)(rows << 16);                            // tensor_dim1[15:0]
  g1[3] = (int)(cols << 16);                            // tile_dim0
  g1[4] = (int)rows;                                    // tile_dim1 (tile_dim2=0)
  g1[5] = (int)cols;                                    // tensor_dim0_stride[31:0]
  i32x4 z4 = (i32x4)0;
#if defined(__clang_major__) && (__clang_major__ >= 23)
  __builtin_amdgcn_tensor_load_to_lds(g0, g1, z4, z4, (i32x8)0, 0);
#else
  __builtin_amdgcn_tensor_load_to_lds(g0, g1, z4, z4, 0);
#endif
}

// ---- WMMA fragment helpers (wave32 layouts, ISA 7.12.2) --------------------

// A fragment: 16x32 bf16 tile, row-major source, row stride lda.
__device__ __forceinline__ v16bf load_A_frag(const bf16_t* __restrict__ src, int lda) {
  int lane = threadIdx.x & 31;
  int m  = lane & 15;
  int kb = (lane >> 4) << 3;       // 0 or 8
  const bf16_t* p = src + m * lda;
  v16bf a;
#pragma unroll
  for (int i = 0; i < 8; ++i) {
    a[i]     = p[kb + i];
    a[8 + i] = p[16 + kb + i];
  }
  return a;
}

// B fragment for out[m][n] = sum_k A[m][k] * W[n][k]; W row-major (row stride ldw).
__device__ __forceinline__ v16bf load_B_fragT(const bf16_t* Wsrc, int ldw) {
  int lane = threadIdx.x & 31;
  int n  = lane & 15;
  int kb = (lane >> 4) << 3;       // 0 or 8
  const bf16_t* p = Wsrc + n * ldw + kb;
  v16bf b;
#pragma unroll
  for (int i = 0; i < 8; ++i) {
    b[i]     = p[i];
    b[8 + i] = p[16 + i];
  }
  return b;
}

__device__ __forceinline__ v8f wmma_bf16(v16bf a, v16bf b, v8f c) {
  return __builtin_amdgcn_wmma_f32_16x16x32_bf16(false, a, false, b, (short)0, c,
                                                 false, false);
}

__device__ __forceinline__ void store_C_f32(float* __restrict__ dst, int ldd, v8f c) {
  int lane = threadIdx.x & 31;
  int col   = lane & 15;
  int rbase = (lane >> 4) << 3;
#pragma unroll
  for (int r = 0; r < 8; ++r) dst[(rbase + r) * ldd + col] = c[r];
}

// ---- K0a: fold weights, bf16 copies, A_basis --------------------------------
__global__ void k0_weights(const float* __restrict__ E_w, const float* __restrict__ M_w,
                           const float* __restrict__ C_w, const float* __restrict__ B_w,
                           const float* __restrict__ D,
                           bf16_t* __restrict__ Wm_bf, bf16_t* __restrict__ Wh_bf,
                           bf16_t* __restrict__ Bw_bf, float* __restrict__ Abasis) {
  int f = blockIdx.x;     // 0..255
  int d = threadIdx.x;    // 0..255
  float wm = 0.f, wh = 0.f;
  for (int e = 0; e < LD; ++e) {
    float ew = E_w[e * LD + d];
    wm = fmaf(M_w[f * LD + e], ew, wm);
    wh = fmaf(C_w[f * LD + e], ew, wh);
  }
  Wm_bf[f * LD + d] = (bf16_t)wm;
  Wh_bf[f * LD + d] = (bf16_t)wh;
  Bw_bf[f * LD + d] = (bf16_t)B_w[f * LD + d];
  if (f < NB) Abasis[f * LD + d] = -(__expf(D[f * LD + d]) + EPSv);
}

// ---- K0b: init vectors ------------------------------------------------------
__global__ void k0_vectors(const float* __restrict__ E_w, const float* __restrict__ init_mean_p,
                           const float* __restrict__ init_log_var, const float* __restrict__ y_log_var,
                           float* __restrict__ initmean0, float* __restrict__ initvar0,
                           float* __restrict__ yvar) {
  int d = threadIdx.x;
  float m0 = 0.f, v0 = 0.f;
  for (int k = 0; k < LD; ++k) {
    float ew = E_w[k * LD + d];
    m0 = fmaf(ew, init_mean_p[k], m0);
    v0 = fmaf(ew * ew, __expf(init_log_var[k]) + EPSv, v0);
  }
  initmean0[d] = m0;
  initvar0[d]  = v0;                  // INIT_SIGMA == 1
  yvar[d]      = __expf(y_log_var[d]) + EPSv;
}

// ---- K0c: Z -> bf16 ---------------------------------------------------------
__global__ void k0_zcvt(const float* __restrict__ Z, bf16_t* __restrict__ Zb) {
  int i = blockIdx.x * blockDim.x + threadIdx.x;   // BT*LD threads
  Zb[i] = (bf16_t)Z[i];
}

// ---- K1: alphas = Z @ B_w^T  (WMMA bf16, B staged via TDM) ------------------
__global__ void k1_alphas(const bf16_t* __restrict__ Zb, const bf16_t* __restrict__ Bwb,
                          float* __restrict__ alphas) {
  __shared__ __align__(16) bf16_t lW[128 * LD];     // 64 KB weight slab
  int wave = threadIdx.x >> 5;
  int rowt = blockIdx.x;                      // 0..2047 (16 rows each)
  int cblk = blockIdx.y;                      // 0..1: cols [cblk*128, +128)
  if (threadIdx.x < 32) {
    tdm_load_2d(lW, Bwb + (size_t)cblk * 128 * LD, 128, LD);
    __builtin_amdgcn_s_wait_tensorcnt(0);
  }
  __syncthreads();
  const bf16_t* Arow = Zb + (size_t)rowt * 16 * LD;
  const bf16_t* Brow = lW + (wave * 16) * LD;       // n_local = wave*16
  v8f c = {};
  for (int k = 0; k < LD; k += 32) {
    __builtin_prefetch(Arow + k + 32, 0, 1);
    v16bf a = load_A_frag(Arow + k, LD);
    v16bf b = load_B_fragT(Brow + k, LD);
    c = wmma_bf16(a, b, c);
  }
  int colt = cblk * 8 + wave;
  store_C_f32(alphas + (size_t)rowt * 16 * LD + colt * 16, LD, c);
}

// ---- K2: logits -> softmax -> A_mat (one wave per row) ----------------------
__global__ void k2_coeff(const float* __restrict__ alphas, const float* __restrict__ coeff_W,
                         const float* __restrict__ coeff_b, const float* __restrict__ Abasis,
                         float* __restrict__ Amat) {
  int wave = threadIdx.x >> 5;
  int lane = threadIdx.x & 31;                // lane == basis index n
  int row  = blockIdx.x * 8 + wave;           // 0..BT-1
  const float* al = alphas  + (size_t)row * LD;
  const float* cw = coeff_W + lane * LD;
  float acc = coeff_b[lane];
  for (int d = 0; d < LD; ++d) acc = fmaf(al[d], cw[d], acc);
  float m = acc;
#pragma unroll
  for (int off = 16; off; off >>= 1) m = fmaxf(m, __shfl_xor(m, off, 32));
  float e = __expf(acc - m);
  float s = e;
#pragma unroll
  for (int off = 16; off; off >>= 1) s += __shfl_xor(s, off, 32);
  float p = e / s;
  float am[8];
#pragma unroll
  for (int j = 0; j < 8; ++j) am[j] = 0.f;
  for (int n = 0; n < NB; ++n) {
    float pn = __shfl(p, n, 32);
    const float* ab = Abasis + n * LD + lane * 8;
#pragma unroll
    for (int j = 0; j < 8; ++j) am[j] = fmaf(pn, ab[j], am[j]);
  }
  float* dst = Amat + (size_t)row * LD + lane * 8;
#pragma unroll
  for (int j = 0; j < 8; ++j) dst[j] = am[j];
}

// ---- scan element ------------------------------------------------------------
__device__ __forceinline__ void scan_elem(float amv, float alv, float tt,
                                          float& eAm, float& eBm, float& eAv, float& eBv) {
  float x   = amv * tt;
  eAm = __expf(x);
  eBm = (eAm - 1.f) / amv * alv;
  eAv = eAm * eAm;                       // exp(2x)
  eBv = 0.5f * (eAv - 1.f) / amv + EPSv; // SIGMA == 1
}

__device__ __forceinline__ void scan_idx(int tid, int& b, int& c, int& d) {
  d = tid & (LD - 1);
  c = (tid >> 8) & (NCH - 1);
  b = tid >> 13;
}

// ---- K3a: per-chunk reduction ------------------------------------------------
__global__ void k3a_chunk(const float* __restrict__ Amat, const float* __restrict__ alphas,
                          const float* __restrict__ obs,
                          float* __restrict__ chAm, float* __restrict__ chBm,
                          float* __restrict__ chAv, float* __restrict__ chBv) {
  int tid = blockIdx.x * blockDim.x + threadIdx.x;  // Bsz*NCH*LD = 131072
  int b, c, d; scan_idx(tid, b, c, d);
  float pAm = 1.f, pBm = 0.f, pAv = 1.f, pBv = 0.f;
  size_t base = ((size_t)b * Tlen + c * CHUNK) * LD + d;
  int tbase = b * Tlen + c * CHUNK;
  for (int t = 0; t < CHUNK; ++t) {
    float eAm, eBm, eAv, eBv;
    scan_elem(Amat[base + (size_t)t * LD], alphas[base + (size_t)t * LD], obs[tbase + t],
              eAm, eBm, eAv, eBv);
    pBm = fmaf(eAm, pBm, eBm); pAm *= eAm;
    pBv = fmaf(eAv, pBv, eBv); pAv *= eAv;
  }
  int ci = (b * LD + d) * NCH + c;
  chAm[ci] = pAm; chBm[ci] = pBm; chAv[ci] = pAv; chBv[ci] = pBv;
}

// ---- K3b: exclusive scan over chunk summaries -------------------------------
__global__ void k3b_scan(float* __restrict__ chAm, float* __restrict__ chBm,
                         float* __restrict__ chAv, float* __restrict__ chBv) {
  int bd = blockIdx.x * blockDim.x + threadIdx.x;   // Bsz*LD = 4096
  float cAm = 1.f, cBm = 0.f, cAv = 1.f, cBv = 0.f;
  for (int c = 0; c < NCH; ++c) {
    int ci = bd * NCH + c;
    float aM = chAm[ci], bM = chBm[ci], aV = chAv[ci], bV = chBv[ci];
    chAm[ci] = cAm; chBm[ci] = cBm; chAv[ci] = cAv; chBv[ci] = cBv;
    cBm = fmaf(aM, cBm, bM); cAm *= aM;
    cBv = fmaf(aV, cBv, bV); cAv *= aV;
  }
}

// ---- K3c: apply prefix, emit U and V = sqrt(var) as bf16 --------------------
__global__ void k3c_apply(const float* __restrict__ Amat, const float* __restrict__ alphas,
                          const float* __restrict__ obs,
                          const float* __restrict__ chAm, const float* __restrict__ chBm,
                          const float* __restrict__ chAv, const float* __restrict__ chBv,
                          const float* __restrict__ initmean0, const float* __restrict__ initvar0,
                          const float* __restrict__ yvar,
                          bf16_t* __restrict__ Ub, bf16_t* __restrict__ Vb) {
  int tid = blockIdx.x * blockDim.x + threadIdx.x;
  int b, c, d; scan_idx(tid, b, c, d);
  int ci = (b * LD + d) * NCH + c;
  float cAm = chAm[ci], cBm = chBm[ci], cAv = chAv[ci], cBv = chBv[ci];
  float im0 = initmean0[d], iv0 = initvar0[d], yv = yvar[d];
  size_t base = ((size_t)b * Tlen + c * CHUNK) * LD + d;
  int tbase = b * Tlen + c * CHUNK;
  for (int t = 0; t < CHUNK; ++t) {
    float eAm, eBm, eAv, eBv;
    scan_elem(Amat[base + (size_t)t * LD], alphas[base + (size_t)t * LD], obs[tbase + t],
              eAm, eBm, eAv, eBv);
    cBm = fmaf(eAm, cBm, eBm); cAm *= eAm;
    cBv = fmaf(eAv, cBv, eBv); cAv *= eAv;
    float U = fmaf(cAm, im0, cBm);                 // xs_mean + init_mean
    float W = fmaf(cAv, iv0, cBv) + yv;            // xs_var + init_var + y_var
    Ub[base + (size_t)t * LD] = (bf16_t)U;
    Vb[base + (size_t)t * LD] = (bf16_t)sqrtf(W);
  }
}

// ---- K4: means = U@Wm^T + Z@Wh^T ; stds = V@Wm^T  (TDM-staged B, 3 streams) -
__global__ void k4_out(const bf16_t* __restrict__ Ub, const bf16_t* __restrict__ Vb,
                       const bf16_t* __restrict__ Zb,
                       const bf16_t* __restrict__ Wm, const bf16_t* __restrict__ Wh,
                       float* __restrict__ means, float* __restrict__ stds) {
  __shared__ __align__(16) bf16_t lWm[128 * LD];    // 64 KB
  __shared__ __align__(16) bf16_t lWh[128 * LD];    // 64 KB
  int wave = threadIdx.x >> 5;
  int rowt = blockIdx.x;
  int cblk = blockIdx.y;                            // cols [cblk*128, +128)
  if (threadIdx.x < 32) {
    tdm_load_2d(lWm, Wm + (size_t)cblk * 128 * LD, 128, LD);
    tdm_load_2d(lWh, Wh + (size_t)cblk * 128 * LD, 128, LD);
    __builtin_amdgcn_s_wait_tensorcnt(0);
  }
  __syncthreads();
  const bf16_t* Au = Ub + (size_t)rowt * 16 * LD;
  const bf16_t* Av = Vb + (size_t)rowt * 16 * LD;
  const bf16_t* Az = Zb + (size_t)rowt * 16 * LD;
  const bf16_t* Bm = lWm + (wave * 16) * LD;
  const bf16_t* Bh = lWh + (wave * 16) * LD;
  v8f cm = {}, cs = {};
  for (int k = 0; k < LD; k += 32) {
    __builtin_prefetch(Au + k + 32, 0, 1);
    __builtin_prefetch(Az + k + 32, 0, 1);
    __builtin_prefetch(Av + k + 32, 0, 1);
    v16bf bm = load_B_fragT(Bm + k, LD);
    v16bf bh = load_B_fragT(Bh + k, LD);
    v16bf au = load_A_frag(Au + k, LD);
    v16bf az = load_A_frag(Az + k, LD);
    v16bf av = load_A_frag(Av + k, LD);
    cm = wmma_bf16(au, bm, cm);
    cm = wmma_bf16(az, bh, cm);
    cs = wmma_bf16(av, bm, cs);
  }
  int colt = cblk * 8 + wave;
  store_C_f32(means + (size_t)rowt * 16 * LD + colt * 16, LD, cm);
  store_C_f32(stds  + (size_t)rowt * 16 * LD + colt * 16, LD, cs);
}

// ---------------------------------------------------------------------------
extern "C" void kernel_launch(void* const* d_in, const int* in_sizes, int n_in,
                              void* d_out, int out_size, void* d_ws, size_t ws_size,
                              hipStream_t stream) {
  (void)in_sizes; (void)n_in; (void)out_size; (void)ws_size;
  const float* Z            = (const float*)d_in[0];
  const float* obs_times    = (const float*)d_in[1];
  const float* D            = (const float*)d_in[2];
  const float* coeff_W      = (const float*)d_in[3];
  const float* coeff_b      = (const float*)d_in[4];
  const float* E_w          = (const float*)d_in[5];
  const float* B_w          = (const float*)d_in[6];
  const float* C_w          = (const float*)d_in[7];
  const float* M_w          = (const float*)d_in[8];
  const float* init_mean_p  = (const float*)d_in[9];
  const float* init_log_var = (const float*)d_in[10];
  const float* y_log_var    = (const float*)d_in[11];

  float* means  = (float*)d_out;
  float* stds   = means + (size_t)BT * LD;
  float* alphas = stds  + (size_t)BT * LD;

  char* w = (char*)d_ws;
  auto take = [&](size_t bytes) -> void* {
    void* p = (void*)w;
    w += (bytes + 255) & ~(size_t)255;
    return p;
  };
  bf16_t* Wm_bf   = (bf16_t*)take((size_t)LD * LD * 2);
  bf16_t* Wh_bf   = (bf16_t*)take((size_t)LD * LD * 2);
  bf16_t* Bw_bf   = (bf16_t*)take((size_t)LD * LD * 2);
  float*  Abasis  = (float*) take((size_t)NB * LD * 4);
  float*  initm0  = (float*) take((size_t)LD * 4);
  float*  initv0  = (float*) take((size_t)LD * 4);
  float*  yvar    = (float*) take((size_t)LD * 4);
  bf16_t* Zb      = (bf16_t*)take((size_t)BT * LD * 2);
  float*  Amat    = (float*) take((size_t)BT * LD * 4);
  bf16_t* Ub      = (bf16_t*)take((size_t)BT * LD * 2);
  bf16_t* Vb      = (bf16_t*)take((size_t)BT * LD * 2);
  float*  chAm    = (float*) take((size_t)Bsz * LD * NCH * 4);
  float*  chBm    = (float*) take((size_t)Bsz * LD * NCH * 4);
  float*  chAv    = (float*) take((size_t)Bsz * LD * NCH * 4);
  float*  chBv    = (float*) take((size_t)Bsz * LD * NCH * 4);

  k0_weights<<<LD, LD, 0, stream>>>(E_w, M_w, C_w, B_w, D, Wm_bf, Wh_bf, Bw_bf, Abasis);
  k0_vectors<<<1, LD, 0, stream>>>(E_w, init_mean_p, init_log_var, y_log_var,
                                   initm0, initv0, yvar);
  k0_zcvt<<<(BT * LD) / 256, 256, 0, stream>>>(Z, Zb);

  k1_alphas<<<dim3(BT / 16, 2), 256, 0, stream>>>(Zb, Bw_bf, alphas);

  k2_coeff<<<BT / 8, 256, 0, stream>>>(alphas, coeff_W, coeff_b, Abasis, Amat);

  const int scanThreads = Bsz * NCH * LD;          // 131072
  k3a_chunk<<<scanThreads / 256, 256, 0, stream>>>(Amat, alphas, obs_times,
                                                   chAm, chBm, chAv, chBv);
  k3b_scan<<<(Bsz * LD) / 256, 256, 0, stream>>>(chAm, chBm, chAv, chBv);
  k3c_apply<<<scanThreads / 256, 256, 0, stream>>>(Amat, alphas, obs_times,
                                                   chAm, chBm, chAv, chBv,
                                                   initm0, initv0, yvar, Ub, Vb);

  k4_out<<<dim3(BT / 16, 2), 256, 0, stream>>>(Ub, Vb, Zb, Wm_bf, Wh_bf, means, stds);
}
